// GMPStack_36532991820473
// MI455X (gfx1250) — matrix operand order repaired and checked by reading, not compiled, not run on hardware
//
#include <hip/hip_runtime.h>

#define NN 50000
#define NE 800000
#define LAT 128
#define NMP 4
#define LN_EPS 1e-5f

typedef unsigned short u16;
typedef __attribute__((ext_vector_type(8)))  float  v8f;
typedef __attribute__((ext_vector_type(16))) __bf16 v16bf;
typedef __attribute__((ext_vector_type(8)))  u16    v8u;
typedef __attribute__((ext_vector_type(16))) u16    v16u;

__device__ __forceinline__ u16 f2bf(float f) {
  unsigned u = __float_as_uint(f);
  u += 0x7FFFu + ((u >> 16) & 1u);          // round-to-nearest-even
  return (u16)(u >> 16);
}

// Async L2->LDS copy, 16B per lane per iteration (ASYNCcnt-tracked, no VGPR
// data path).  nU4 must be a multiple of 256 (all lanes active, EXEC all-1).
// LDS offset = low 32 bits of the flat shared-memory address (LDS aperture
// maps addr[31:0] directly onto the LDS offset).
__device__ __forceinline__ void async_copy_b128(const void* gsrc, void* lds,
                                                int nU4, int tid) {
  const unsigned ldsOff = (unsigned)(uintptr_t)lds;
  for (int t = tid; t < nU4; t += 256) {
    const unsigned voff  = (unsigned)t * 16u;
    const unsigned laddr = ldsOff + voff;
    asm volatile("global_load_async_to_lds_b128 %0, %1, %2"
                 :: "v"(laddr), "v"(voff), "s"(gsrc) : "memory");
  }
}
__device__ __forceinline__ void async_wait0() {
  asm volatile("s_wait_asynccnt 0" ::: "memory");
}

// One 16x16 output tile of a (16 x K) @ (K x 16) GEMM, bf16 inputs, f32 accum.
// A: row-major [16][lda] bf16 in LDS.  Bn: this tile's column block, [16 cols][K]
// (weights pre-transposed to [N][K]); caller passes Bn = W + n*ldb.
// Per-lane fragment layout per CDNA5 ISA 7.12.2.
__device__ __forceinline__ v8f bf16_gemm16(const u16* A, int lda, const u16* Bn,
                                           int K, int lane) {
  const int row = lane & 15;
  const int aK  = (lane & 16) ? 8 : 0;    // A: lanes16-31 hold K+8..15 / K+24..31
  const int bK  = (lane & 16) ? 8 : 0;    // B: lanes16-31 hold K+8..15 / K+24..31
  v8f acc = {};
  for (int kc = 0; kc < K; kc += 32) {
    const u16* pa = A + row * lda + kc + aK;
    v8u a0 = *(const v8u*)pa;             // K = kc+aK+0..7      -> VGPR0..3
    v8u a1 = *(const v8u*)(pa + 16);      // K = kc+16+aK+0..7   -> VGPR4..7
    v16u au = __builtin_shufflevector(a0, a1, 0,1,2,3,4,5,6,7,8,9,10,11,12,13,14,15);
    const u16* pb = Bn + kc + bK;
    v8u b0v = *(const v8u*)pb;            // K = kc+bK+0..7
    v8u b1v = *(const v8u*)(pb + 16);     // K = kc+16+bK+0..7
    v16u bu = __builtin_shufflevector(b0v, b1v, 0,1,2,3,4,5,6,7,8,9,10,11,12,13,14,15);
    acc = __builtin_amdgcn_wmma_f32_16x16x32_bf16(
        false, __builtin_bit_cast(v16bf, au),
        false, __builtin_bit_cast(v16bf, bu),
        (short)0, acc, false, false);
  }
  return acc;
}

// -------- weight pre-transpose: f32 [K][128] -> bf16 [128][KP], zero-pad K..KP
__global__ void wt_kernel(const float* __restrict__ src, u16* __restrict__ dst,
                          int K, int KP) {
  int idx = blockIdx.x * 256 + threadIdx.x;
  if (idx >= 128 * KP) return;
  int n = idx / KP, k = idx % KP;
  dst[idx] = (k < K) ? f2bf(src[k * 128 + n]) : (u16)0;
}

__global__ void cvt_kernel(const float* __restrict__ src, u16* __restrict__ dst, int n) {
  int i = blockIdx.x * 256 + threadIdx.x;
  if (i < n) dst[i] = f2bf(src[i]);
}

__global__ void zero_kernel(float* __restrict__ p, int n) {
  int i = blockIdx.x * 256 + threadIdx.x;
  if (i < n) p[i] = 0.f;
}

// -------- edge MLP + LayerNorm + scatter-add (segment_sum by j) -------------
// block = 256 threads = 8 waves; 256 edges per block in 16 M-tiles of 16.
__global__ __launch_bounds__(256)
void edge_kernel(const int* __restrict__ g, const float* __restrict__ pos,
                 const u16* __restrict__ xbf,
                 const u16* __restrict__ gW0, const u16* __restrict__ gW1,
                 const u16* __restrict__ gW2,
                 const float* __restrict__ pb0, const float* __restrict__ pb1,
                 const float* __restrict__ pb2,
                 const float* __restrict__ pgam, const float* __restrict__ pbet,
                 float* __restrict__ aggr) {
  __shared__ __align__(16) u16 sW0[128 * 288];
  __shared__ __align__(16) u16 sW1[128 * 128];
  __shared__ __align__(16) u16 sW2[128 * 128];
  __shared__ __align__(16) union { u16 a0[16 * 288]; float h[16 * 128]; } sU;
  __shared__ __align__(16) u16 sA[16 * 128];     // reused for A1 and A2
  __shared__ float sMU[16], sRS[16];

  const int tid = threadIdx.x;
  // stage weights L2 -> LDS via async-to-LDS (ASYNCcnt), no VGPR data path
  async_copy_b128(gW0, sW0, 128 * 288 / 8, tid);
  async_copy_b128(gW1, sW1, 128 * 128 / 8, tid);
  async_copy_b128(gW2, sW2, 128 * 128 / 8, tid);
  async_wait0();
  __syncthreads();

  const int wv = tid >> 5, lane = tid & 31;
  const int n  = (wv << 4) | (lane & 15);     // output column this lane owns
  const int mh = (lane & 16) ? 8 : 0;         // D rows: r + mh

  for (int mt = 0; mt < 16; ++mt) {
    const int e0 = blockIdx.x * 256 + mt * 16;
    // build A0 = [dx,dy,dz,|d| , x_i(128) , x_j(128), pad..288] in bf16
    for (int idx = tid; idx < 16 * 288; idx += 256) {
      const int el = idx / 288, k = idx % 288;
      const int e  = e0 + el;
      const int vi = g[e], vj = g[NE + e];
      u16 val;
      if (k < 4) {
        const float dx = pos[vi * 3 + 0] - pos[vj * 3 + 0];
        const float dy = pos[vi * 3 + 1] - pos[vj * 3 + 1];
        const float dz = pos[vi * 3 + 2] - pos[vj * 3 + 2];
        float f = (k == 0) ? dx : (k == 1) ? dy : (k == 2) ? dz
                  : sqrtf(dx * dx + dy * dy + dz * dz);
        val = f2bf(f);
      } else if (k < 132) {
        val = xbf[vi * LAT + (k - 4)];
      } else if (k < 260) {
        val = xbf[vj * LAT + (k - 132)];
      } else {
        val = 0;
      }
      sU.a0[idx] = val;
    }
    __syncthreads();

    v8f h = bf16_gemm16(sU.a0, 288, sW0 + n * 288, 288, lane);
    {
      const float bb = pb0[n];
      #pragma unroll
      for (int r = 0; r < 8; ++r) sA[(r + mh) * LAT + n] = f2bf(fmaxf(h[r] + bb, 0.f));
    }
    __syncthreads();
    h = bf16_gemm16(sA, 128, sW1 + n * 128, 128, lane);
    __syncthreads();                           // all waves done reading A1
    {
      const float bb = pb1[n];
      #pragma unroll
      for (int r = 0; r < 8; ++r) sA[(r + mh) * LAT + n] = f2bf(fmaxf(h[r] + bb, 0.f));
    }
    __syncthreads();
    h = bf16_gemm16(sA, 128, sW2 + n * 128, 128, lane);
    {
      const float bb = pb2[n];
      #pragma unroll
      for (int r = 0; r < 8; ++r) sU.h[(r + mh) * LAT + n] = h[r] + bb;
    }
    __syncthreads();
    if (tid < 16) {                            // LayerNorm stats per edge-row
      float s = 0.f, s2 = 0.f;
      for (int c = 0; c < LAT; ++c) { float v = sU.h[tid * LAT + c]; s += v; s2 += v * v; }
      const float mu = s * (1.f / LAT);
      sMU[tid] = mu;
      sRS[tid] = rsqrtf(fmaxf(s2 * (1.f / LAT) - mu * mu, 0.f) + LN_EPS);
    }
    __syncthreads();
    for (int idx = tid; idx < 16 * LAT; idx += 256) {
      const int m = idx >> 7, c = idx & 127;
      const int vj = g[NE + e0 + m];
      const float y = (sU.h[idx] - sMU[m]) * sRS[m] * pgam[c] + pbet[c];
      atomicAdd(&aggr[vj * LAT + c], y);
    }
    __syncthreads();
  }
}

// -------- node MLP + LayerNorm + residual -----------------------------------
__global__ __launch_bounds__(256)
void node_kernel(const float* __restrict__ xcur, const u16* __restrict__ xbf,
                 const float* __restrict__ aggr,
                 const u16* __restrict__ gW0, const u16* __restrict__ gW1,
                 const u16* __restrict__ gW2,
                 const float* __restrict__ pb0, const float* __restrict__ pb1,
                 const float* __restrict__ pb2,
                 const float* __restrict__ pgam, const float* __restrict__ pbet,
                 float* __restrict__ xout) {
  __shared__ __align__(16) u16 sW0[128 * 256];
  __shared__ __align__(16) u16 sW1[128 * 128];
  __shared__ __align__(16) u16 sW2[128 * 128];
  __shared__ __align__(16) union { u16 a0[16 * 256]; float h[16 * 128]; } sU;
  __shared__ __align__(16) u16 sA[16 * 128];
  __shared__ float sMU[16], sRS[16];

  const int tid = threadIdx.x;
  async_copy_b128(gW0, sW0, 128 * 256 / 8, tid);
  async_copy_b128(gW1, sW1, 128 * 128 / 8, tid);
  async_copy_b128(gW2, sW2, 128 * 128 / 8, tid);
  async_wait0();
  __syncthreads();

  const int wv = tid >> 5, lane = tid & 31;
  const int n  = (wv << 4) | (lane & 15);
  const int mh = (lane & 16) ? 8 : 0;

  for (int mt = 0; mt < 16; ++mt) {
    const int n0 = blockIdx.x * 256 + mt * 16;
    if (n0 >= NN) break;                       // uniform across block (NN % 16 == 0)
    for (int idx = tid; idx < 16 * 256; idx += 256) {
      const int nl = idx >> 8, k = idx & 255;
      const int node = n0 + nl;
      sU.a0[idx] = (k < 128) ? xbf[node * LAT + k] : f2bf(aggr[node * LAT + (k - 128)]);
    }
    __syncthreads();

    v8f h = bf16_gemm16(sU.a0, 256, sW0 + n * 256, 256, lane);
    {
      const float bb = pb0[n];
      #pragma unroll
      for (int r = 0; r < 8; ++r) sA[(r + mh) * LAT + n] = f2bf(fmaxf(h[r] + bb, 0.f));
    }
    __syncthreads();
    h = bf16_gemm16(sA, 128, sW1 + n * 128, 128, lane);
    __syncthreads();
    {
      const float bb = pb1[n];
      #pragma unroll
      for (int r = 0; r < 8; ++r) sA[(r + mh) * LAT + n] = f2bf(fmaxf(h[r] + bb, 0.f));
    }
    __syncthreads();
    h = bf16_gemm16(sA, 128, sW2 + n * 128, 128, lane);
    {
      const float bb = pb2[n];
      #pragma unroll
      for (int r = 0; r < 8; ++r) sU.h[(r + mh) * LAT + n] = h[r] + bb;
    }
    __syncthreads();
    if (tid < 16) {
      float s = 0.f, s2 = 0.f;
      for (int c = 0; c < LAT; ++c) { float v = sU.h[tid * LAT + c]; s += v; s2 += v * v; }
      const float mu = s * (1.f / LAT);
      sMU[tid] = mu;
      sRS[tid] = rsqrtf(fmaxf(s2 * (1.f / LAT) - mu * mu, 0.f) + LN_EPS);
    }
    __syncthreads();
    for (int idx = tid; idx < 16 * LAT; idx += 256) {
      const int m = idx >> 7, c = idx & 127;
      const int node = n0 + m;
      xout[node * LAT + c] =
          (sU.h[idx] - sMU[m]) * sRS[m] * pgam[c] + pbet[c] + xcur[node * LAT + c];
    }
    __syncthreads();
  }
}

extern "C" void kernel_launch(void* const* d_in, const int* in_sizes, int n_in,
                              void* d_out, int out_size, void* d_ws, size_t ws_size,
                              hipStream_t stream) {
  (void)out_size; (void)ws_size;
  // ---- locate x, g, pos, params (robust to insertion-order vs sorted flatten)
  const float* x_in; const int* g; const float* pos; int pstart;
  if (in_sizes[0] == NN * LAT) {               // insertion order: x, g, pos, params
    x_in = (const float*)d_in[0]; g = (const int*)d_in[1];
    pos  = (const float*)d_in[2]; pstart = 3;
  } else {                                     // sorted: g, params..., pos, x
    g = (const int*)d_in[0]; pstart = 1;
    pos  = (const float*)d_in[n_in - 2];
    x_in = (const float*)d_in[n_in - 1];
  }

  struct Mlp { const float *W0, *b0, *W1, *b1, *W2, *b2, *gamma, *beta; };
  Mlp mlp[8];                                  // [2l]=edge, [2l+1]=node
  int c = pstart;
  for (int m = 0; m < 8; ++m) {
    const float* a[8];
    for (int k = 0; k < 8; ++k) a[k] = (const float*)d_in[c + k];
    if (in_sizes[c] > 128) {  // W0,b0,W1,b1,W2,b2,gamma,beta
      mlp[m] = { a[0], a[1], a[2], a[3], a[4], a[5], a[6], a[7] };
    } else {                   // beta,gamma,W0,b0,W1,b1,W2,b2 (jax sorted keys)
      mlp[m] = { a[2], a[3], a[4], a[5], a[6], a[7], a[1], a[0] };
    }
    c += 8;
  }

  // ---- workspace carve-up (all offsets 256B aligned)
  char* ws = (char*)d_ws;
  u16*   xbf  = (u16*)ws;   ws += (size_t)NN * LAT * 2;   // bf16 copy of current x
  float* wsx  = (float*)ws; ws += (size_t)NN * LAT * 4;   // x ping buffer
  float* aggr = (float*)ws; ws += (size_t)NN * LAT * 4;   // segment sums
  u16* wb[8][3];
  for (int m = 0; m < 8; ++m) {
    const int KP0 = (m & 1) ? 256 : 288;
    wb[m][0] = (u16*)ws; ws += (size_t)128 * KP0 * 2;
    wb[m][1] = (u16*)ws; ws += (size_t)128 * 128 * 2;
    wb[m][2] = (u16*)ws; ws += (size_t)128 * 128 * 2;
  }

  // ---- pre-transpose weights to bf16 [N][K] with K zero-padding
  for (int m = 0; m < 8; ++m) {
    const int K0 = (m & 1) ? 256 : 260, KP0 = (m & 1) ? 256 : 288;
    wt_kernel<<<(128 * KP0 + 255) / 256, 256, 0, stream>>>(mlp[m].W0, wb[m][0], K0, KP0);
    wt_kernel<<<(128 * 128 + 255) / 256, 256, 0, stream>>>(mlp[m].W1, wb[m][1], 128, 128);
    wt_kernel<<<(128 * 128 + 255) / 256, 256, 0, stream>>>(mlp[m].W2, wb[m][2], 128, 128);
  }

  const float* xc = x_in;
  cvt_kernel<<<(NN * LAT + 255) / 256, 256, 0, stream>>>(xc, xbf, NN * LAT);

  for (int l = 0; l < NMP; ++l) {
    float* xo = (l & 1) ? (float*)d_out : wsx;  // l0->ws, l1->out, l2->ws, l3->out
    zero_kernel<<<(NN * LAT + 255) / 256, 256, 0, stream>>>(aggr, NN * LAT);
    const Mlp& E = mlp[2 * l]; const Mlp& Nd = mlp[2 * l + 1];
    edge_kernel<<<NE / 256, 256, 0, stream>>>(
        g, pos, xbf, wb[2 * l][0], wb[2 * l][1], wb[2 * l][2],
        E.b0, E.b1, E.b2, E.gamma, E.beta, aggr);
    node_kernel<<<(NN + 255) / 256, 256, 0, stream>>>(
        xc, xbf, aggr, wb[2 * l + 1][0], wb[2 * l + 1][1], wb[2 * l + 1][2],
        Nd.b0, Nd.b1, Nd.b2, Nd.gamma, Nd.beta, xo);
    if (l < NMP - 1)
      cvt_kernel<<<(NN * LAT + 255) / 256, 256, 0, stream>>>(xo, xbf, NN * LAT);
    xc = xo;
  }
}
